// NTXentLossChildren_85624468013603
// MI455X (gfx1250) — matrix operand order repaired and checked by reading, not compiled, not run on hardware
//
#include <hip/hip_runtime.h>
#include <hip/hip_bf16.h>
#include <math.h>

// NT-Xent children loss for MI455X (gfx1250, wave32, WMMA).
// B=64 anchors, C=32 children, D=256 features.
//
// Heavy compute = two 2048x2048x256 gram matrices, done blockwise with
// v_wmma_f32_16x16x32_f16 using split-f16 inputs (hi+lo, Markidis scheme:
// A.B^T ~= AhBh + AhBl + AlBh) for ~fp32 accuracy. Norms / logsumexp in fp32.

#define NB 64          // batch
#define NC 32          // children
#define ND 256         // feature dim
#define MCH 8          // m-values per gram block
#define CHUNKS (NB / MCH)
#define EPSV 1e-8f
#define TEMPV 0.1f

typedef __attribute__((ext_vector_type(16))) _Float16 v16h;
typedef __attribute__((ext_vector_type(8)))  _Float16 v8h;
typedef __attribute__((ext_vector_type(8)))  float    v8f;

__device__ __forceinline__ float wave_sum(float v) {
  for (int off = 16; off >= 1; off >>= 1) v += __shfl_xor(v, off, 32);
  return v;
}
__device__ __forceinline__ float wave_max(float v) {
  for (int off = 16; off >= 1; off >>= 1) v = fmaxf(v, __shfl_xor(v, off, 32));
  return v;
}

// ---------------------------------------------------------------------------
// Kernel 1: fp32 -> split f16 (hi+lo), per-row squared norms, per-row pos dots.
// One block per row (B*C = 2048 rows), 256 threads = 1 element/thread.
// ---------------------------------------------------------------------------
__global__ __launch_bounds__(256) void k_prep(
    const float* __restrict__ CI, const float* __restrict__ CJ,
    _Float16* __restrict__ AhH, _Float16* __restrict__ AhL,
    _Float16* __restrict__ BhH, _Float16* __restrict__ BhL,
    float* __restrict__ rn_i, float* __restrict__ rn_j,
    float* __restrict__ rowdot) {
  const int row = blockIdx.x;
  const int t   = threadIdx.x;
  const int g   = row * ND + t;
  const float a = CI[g];
  const float b = CJ[g];
  const _Float16 ah = (_Float16)a;
  const _Float16 bh = (_Float16)b;
  AhH[g] = ah; AhL[g] = (_Float16)(a - (float)ah);
  BhH[g] = bh; BhL[g] = (_Float16)(b - (float)bh);

  float sa = a * a, sb = b * b, sab = a * b;
  const int lane = t & 31, wv = t >> 5;
  sa = wave_sum(sa); sb = wave_sum(sb); sab = wave_sum(sab);
  __shared__ float red[8][3];
  if (lane == 0) { red[wv][0] = sa; red[wv][1] = sb; red[wv][2] = sab; }
  __syncthreads();
  if (t == 0) {
    float A = 0.f, Bv = 0.f, Cv = 0.f;
    for (int w = 0; w < 8; ++w) { A += red[w][0]; Bv += red[w][1]; Cv += red[w][2]; }
    rn_i[row] = A; rn_j[row] = Bv; rowdot[row] = Cv;
  }
}

// ---------------------------------------------------------------------------
// Kernel 2: gram blocks via WMMA + in-register index gather.
// grid = NB*CHUNKS = 512 blocks (anchor i x chunk of 8 m-values), 8 waves each.
// wave -> (gram = w>>2, tm = (w>>1)&1, tn = w&1):
//   gram 0: children_is[i] . children_js[m]^T  (for sim_j / neg_idx_j)
//   gram 1: children_is[i] . children_is[m]^T  (for sim_k / neg_idx_k)
// Fragment packing (contiguous k order, same in A and B):
//   A 16x32 f16: lane -> M=lane&15, K = k0 + 16*(lane>>4) .. +15
//                (matches B). Two contiguous 16B loads/lane.
//   B 32x16 f16: lane -> N=lane&15, K = k0 + 16*(lane>>4) .. +15. One 32B load.
//   C/D 16x16 f32: lane holds N=lane&15; VGPR v holds M = v + 8*(lane>>4).
// ---------------------------------------------------------------------------
__global__ __launch_bounds__(256) void k_gram(
    const _Float16* __restrict__ AhH, const _Float16* __restrict__ AhL,
    const _Float16* __restrict__ BhH, const _Float16* __restrict__ BhL,
    const int* __restrict__ idx_j, const int* __restrict__ idx_k,
    float* __restrict__ dot_j, float* __restrict__ dot_k) {
  __shared__ float accJ[MCH], accK[MCH];
  const int i   = blockIdx.x / CHUNKS;
  const int m0  = (blockIdx.x % CHUNKS) * MCH;
  const int tid = threadIdx.x;
  if (tid < MCH) accJ[tid] = 0.f;
  else if (tid < 2 * MCH) accK[tid - MCH] = 0.f;
  __syncthreads();

  const int lane = tid & 31;
  const int wave = tid >> 5;
  const int gram = wave >> 2;
  const int tm   = (wave >> 1) & 1;
  const int tn   = wave & 1;
  const int g    = lane >> 4;    // half-wave select
  const int nrow = lane & 15;    // A row within tile / B column within tile

  // A fragments (hi and lo) for rows 16*tm.. of anchor i, K=256 (8 chunks).
  v16h afH[8], afL[8];
  const size_t aoff = (size_t)(i * NC + tm * 16 + nrow) * ND;
  #pragma unroll
  for (int ko = 0; ko < 8; ++ko) {
    const _Float16* pH = AhH + aoff + ko * 32 + g * 16;
    const _Float16* pL = AhL + aoff + ko * 32 + g * 16;
    v8h loH = *(const v8h*)(pH);
    v8h hiH = *(const v8h*)(pH + 8);
    v8h loL = *(const v8h*)(pL);
    v8h hiL = *(const v8h*)(pL + 8);
    v16h aH, aL;
    #pragma unroll
    for (int e = 0; e < 8; ++e) {
      aH[e] = loH[e]; aH[e + 8] = hiH[e];
      aL[e] = loL[e]; aL[e + 8] = hiL[e];
    }
    afH[ko] = aH; afL[ko] = aL;
  }

  const _Float16* BsrcH = gram ? AhH   : BhH;
  const _Float16* BsrcL = gram ? AhL   : BhL;
  const int*      isrc  = gram ? idx_k : idx_j;
  float*          accS  = gram ? accK  : accJ;

  for (int m = m0; m < m0 + MCH; ++m) {
    const size_t boff = (size_t)(m * NC + tn * 16 + nrow) * ND + g * 16;
    const _Float16* BrowH = BsrcH + boff;
    const _Float16* BrowL = BsrcL + boff;
    if (m < m0 + MCH - 1) {                       // next m's rows
      __builtin_prefetch(BrowH + NC * ND, 0, 1);  // global_prefetch_b8
      __builtin_prefetch(BrowL + NC * ND, 0, 1);
    }

    v8f acc = {};
    #pragma unroll
    for (int ko = 0; ko < 8; ++ko) {
      v16h bH = *(const v16h*)(BrowH + ko * 32);
      v16h bL = *(const v16h*)(BrowL + ko * 32);
      acc = __builtin_amdgcn_wmma_f32_16x16x32_f16(
          false, afH[ko], false, bH, (short)0, acc, false, false);
      acc = __builtin_amdgcn_wmma_f32_16x16x32_f16(
          false, afH[ko], false, bL, (short)0, acc, false, false);
      acc = __builtin_amdgcn_wmma_f32_16x16x32_f16(
          false, afL[ko], false, bH, (short)0, acc, false, false);
    }

    // Gather: for each row c in this tile, pick column idx[c] if it lies in
    // this tile's column range. Each D element lives in exactly one lane.
    const int* idx = isrc + (size_t)(i * NB + m) * NC;
    float s = 0.f;
    #pragma unroll
    for (int v = 0; v < 8; ++v) {
      const int M   = v + 8 * g;              // row within 16x16 tile
      const int col = idx[tm * 16 + M];       // 0..31 child index
      if ((col >> 4) == tn && (col & 15) == nrow) s += acc[v];
    }
    s = wave_sum(s);
    if (lane == 0) atomicAdd(&accS[m - m0], s);   // ds_add_f32
  }
  __syncthreads();
  if (tid < MCH)          dot_j[i * NB + m0 + tid]       = accJ[tid];
  else if (tid < 2 * MCH) dot_k[i * NB + m0 + tid - MCH] = accK[tid - MCH];
}

// ---------------------------------------------------------------------------
// Kernel 3: per-anchor logsumexp. grid = 64, block = 128 (4 waves).
// threads 0..63: sim_j[m]; threads 64..127: sim_k[m].
// ---------------------------------------------------------------------------
__global__ __launch_bounds__(128) void k_finalize(
    const int* __restrict__ idx_j, const int* __restrict__ idx_k,
    const float* __restrict__ rn_i, const float* __restrict__ rn_j,
    const float* __restrict__ rowdot,
    const float* __restrict__ dot_j, const float* __restrict__ dot_k,
    float* __restrict__ per_i) {
  const int i = blockIdx.x;
  const int t = threadIdx.x;
  const int lane = t & 31, wv = t >> 5;
  const int m = t & 63;

  float na2 = 0.f, nb2 = 0.f, posd = 0.f;
  for (int c = 0; c < NC; ++c) {
    na2 += rn_i[i * NC + c];
    nb2 += rn_j[i * NC + c];
    posd += rowdot[i * NC + c];
  }
  const float na = sqrtf(na2), nbn = sqrtf(nb2);
  const float pos = posd / fmaxf(na * nbn, EPSV) / TEMPV;

  float logit;
  if (t < 64) {
    const int* idx = idx_j + (size_t)(i * NB + m) * NC;
    float g2 = 0.f;
    for (int c = 0; c < NC; ++c) g2 += rn_j[m * NC + idx[c]];
    logit = (m == i) ? -__builtin_inff()
                     : dot_j[i * NB + m] / fmaxf(na * sqrtf(g2), EPSV) / TEMPV;
  } else {
    const int* idx = idx_k + (size_t)(i * NB + m) * NC;
    float g2 = 0.f;
    for (int c = 0; c < NC; ++c) g2 += rn_i[m * NC + idx[c]];
    logit = (m == i) ? -__builtin_inff()
                     : dot_k[i * NB + m] / fmaxf(na * sqrtf(g2), EPSV) / TEMPV;
  }

  __shared__ float redm[4], reds[4];
  float mx = wave_max(logit);
  if (lane == 0) redm[wv] = mx;
  __syncthreads();
  mx = fmaxf(fmaxf(redm[0], redm[1]), fmaxf(redm[2], redm[3]));
  mx = fmaxf(mx, pos);

  float e = wave_sum(expf(logit - mx));   // exp(-inf)=0 drops diagonal
  if (lane == 0) reds[wv] = e;
  __syncthreads();
  if (t == 0) {
    const float tot = reds[0] + reds[1] + reds[2] + reds[3] + expf(pos - mx);
    per_i[i] = mx + logf(tot) - pos;      // lse_i - pos_i
  }
}

// ---------------------------------------------------------------------------
// Kernel 4: final scalar. 1 block, 64 threads (2 waves).
// ---------------------------------------------------------------------------
__global__ __launch_bounds__(64) void k_sum(const float* __restrict__ per_i,
                                            float* __restrict__ out) {
  const int t = threadIdx.x;
  float v = wave_sum(per_i[t]);
  __shared__ float red[2];
  if ((t & 31) == 0) red[t >> 5] = v;
  __syncthreads();
  if (t == 0) out[0] = (red[0] + red[1]) / (2.0f * NB);
}

// ---------------------------------------------------------------------------
extern "C" void kernel_launch(void* const* d_in, const int* in_sizes, int n_in,
                              void* d_out, int out_size, void* d_ws, size_t ws_size,
                              hipStream_t stream) {
  (void)in_sizes; (void)n_in; (void)out_size; (void)ws_size;
  const float* CI  = (const float*)d_in[0];   // children_is [64,32,256] f32
  const float* CJ  = (const float*)d_in[1];   // children_js [64,32,256] f32
  const int*   IJ  = (const int*)d_in[2];     // neg_idx_j   [64,64,32]  i32
  const int*   IK  = (const int*)d_in[3];     // neg_idx_k   [64,64,32]  i32
  // d_in[4] partnet_ids unused (reference masks with eye(B) directly).
  float* out = (float*)d_out;

  // Workspace layout (bytes); total ~4.3 MB.
  char* ws = (char*)d_ws;
  _Float16* AhH    = (_Float16*)(ws + 0 * (1u << 20));  // 1 MiB each
  _Float16* AhL    = (_Float16*)(ws + 1 * (1u << 20));
  _Float16* BhH    = (_Float16*)(ws + 2 * (1u << 20));
  _Float16* BhL    = (_Float16*)(ws + 3 * (1u << 20));
  float*    rn_i   = (float*)(ws + (4u << 20));                 // 8 KiB
  float*    rn_j   = (float*)(ws + (4u << 20) + 8192);          // 8 KiB
  float*    rowdot = (float*)(ws + (4u << 20) + 16384);         // 8 KiB
  float*    dot_j  = (float*)(ws + (4u << 20) + 24576);         // 16 KiB
  float*    dot_k  = (float*)(ws + (4u << 20) + 40960);         // 16 KiB
  float*    per_i  = (float*)(ws + (4u << 20) + 57344);         // 256 B

  k_prep<<<NB * NC, 256, 0, stream>>>(CI, CJ, AhH, AhL, BhH, BhL, rn_i, rn_j, rowdot);
  k_gram<<<NB * CHUNKS, 256, 0, stream>>>(AhH, AhL, BhH, BhL, IJ, IK, dot_j, dot_k);
  k_finalize<<<NB, 128, 0, stream>>>(IJ, IK, rn_i, rn_j, rowdot, dot_j, dot_k, per_i);
  k_sum<<<1, 64, 0, stream>>>(per_i, out);
}
